// GraphModel_75960791597213
// MI455X (gfx1250) — matrix operand order repaired and checked by reading, not compiled, not run on hardware
//
#include <hip/hip_runtime.h>
#include <hip/hip_bf16.h>
#include <cstdint>
#include <cstddef>

// ---------------------------------------------------------------------------
// GConvGRU (ChebConv K=3) for MI455X / gfx1250.
//  - Dense Chebyshev GEMMs -> v_wmma_f32_16x16x32_f16, weights staged in LDS
//    (TDM tensor_load_to_lds when available, else cooperative copy)
//  - Sparse propagation    -> L2-resident gather + global_atomic_add_f32
//  - GRU recurrence        -> host-side kernel sequence on `stream`
// ---------------------------------------------------------------------------

typedef __attribute__((ext_vector_type(16))) _Float16 v16h;
typedef __attribute__((ext_vector_type(8)))  float    v8f;
typedef unsigned int u32x4 __attribute__((ext_vector_type(4)));
typedef int          i32x8 __attribute__((ext_vector_type(8)));
typedef int          i32x4 __attribute__((ext_vector_type(4)));

#if defined(__has_builtin)
#if __has_builtin(__builtin_amdgcn_tensor_load_to_lds) && \
    __has_builtin(__builtin_amdgcn_s_wait_tensorcnt)
#define HAVE_TDM 1
#endif
#endif

#define N_NODES 20000
#define N_EDGES 640000
#define N_B     4
#define N_T     10
#define HIDF    64
#define SLAB_H  (3 * 64 * 64)           // halves per gate weight slab
#define GEMM_BLOCKS 80

static inline int cdiv(int a, int b) { return (a + b - 1) / b; }

// ------------------------------ graph norm ---------------------------------

__global__ void deg_kernel(const int* __restrict__ ei, const float* __restrict__ w,
                           float* __restrict__ deg, int E) {
  int e = blockIdx.x * blockDim.x + threadIdx.x;
  if (e < E) atomicAdd(&deg[ei[e]], w[e]);
}

__global__ void norm_kernel(const int* __restrict__ ei, const float* __restrict__ w,
                            const float* __restrict__ deg, float* __restrict__ nrm, int E) {
  int e = blockIdx.x * blockDim.x + threadIdx.x;
  if (e >= E) return;
  int r = ei[e], c = ei[E + e];
  float dr = deg[r], dc = deg[c];
  float ir = dr > 0.f ? rsqrtf(dr) : 0.f;
  float ic = dc > 0.f ? rsqrtf(dc) : 0.f;
  nrm[e] = -ir * w[e] * ic;
}

// --------------------- weight f32 -> f16, WMMA-B layout --------------------
// Input : W  f32 [3][fin][64]   (K, kk, col)
// Output: Wh f16 [3][64][64]    (K-matrix, col, kk)  zero-padded kk to 64.
// Lane n of the wave later reads 16 contiguous halves Wh[k][col][koff..koff+15].

__global__ void prep_w_kernel(const float* __restrict__ W, _Float16* __restrict__ Wh, int fin) {
  int gid = blockIdx.x * blockDim.x + threadIdx.x;       // 3*64*64
  if (gid >= 3 * 64 * 64) return;
  int k   = gid >> 12;
  int r   = gid & 4095;
  int col = r >> 6;
  int kk  = r & 63;
  float v = (kk < fin) ? W[((size_t)k * fin + kk) * 64 + col] : 0.f;
  Wh[((size_t)k * 64 + col) * 64 + kk] = (_Float16)v;
}

// ------------------------- sparse propagation ------------------------------
// out[col[e]][:] += nrm[e] * x[row[e]][:]   (x is L2-resident: 5 MB)

__global__ void prop_kernel(const float* __restrict__ x, const int* __restrict__ ei,
                            const float* __restrict__ nrm, float* __restrict__ out,
                            int E, int F) {
  long long gid = (long long)blockIdx.x * blockDim.x + threadIdx.x;
  int pairs = F >> 1;
  long long e = gid / pairs;
  int p = (int)(gid - e * pairs);
  if (e >= E) return;
  int r = ei[e], c = ei[E + e];
  float nv = nrm[e];
  const float2 v = *(const float2*)(x + (size_t)r * F + 2 * p);
  float* o = out + (size_t)c * F + 2 * p;
  atomicAdd(o,     nv * v.x);
  atomicAdd(o + 1, nv * v.y);
}

// Tx2 = 2*prop(Tx1) - Tx0
__global__ void fixup_kernel(float* __restrict__ p2, const float* __restrict__ x0, int n) {
  int i = blockIdx.x * blockDim.x + threadIdx.x;
  if (i < n) p2[i] = 2.f * p2[i] - x0[i];
}

// --------------------------- WMMA Chebyshev GEMM ---------------------------
// One wave computes 16-row x 64-col tiles (grid-stride) for up to NOUT gates:
//   O[o] (+)= A0 @ W[o][0] + A1 @ W[o][1] + A2 @ W[o][2] + b[o]
// A_k: f32 [N][Kdim] (Kdim = 8 or 64). Weights live in LDS (staged by TDM or
// cooperative copy) in the WMMA-B lane layout produced by prep_w_kernel.

template <int NOUT>
__global__ __launch_bounds__(256)
void cheb_gemm_kernel(const float* __restrict__ A0, const float* __restrict__ A1,
                      const float* __restrict__ A2,
                      const _Float16* __restrict__ W0, const _Float16* __restrict__ W1,
                      const float* __restrict__ b0, const float* __restrict__ b1,
                      float* __restrict__ O0, float* __restrict__ O1,
                      int nNodes, int Kdim, int kTiles, int accum) {
  extern __shared__ _Float16 sW[];                 // NOUT * SLAB_H halves
  const _Float16* Wsrc[2] = {W0, W1};

  // ---- stage weight slabs into LDS ----
#if defined(HAVE_TDM)
  if (threadIdx.x < 32) {                          // wave 0 issues TDM DMA
#pragma unroll
    for (int o = 0; o < NOUT; ++o) {
      const unsigned n = SLAB_H;                   // elements (2B each)
      unsigned lds_addr = (unsigned)(uintptr_t)(sW + (size_t)o * SLAB_H);
      unsigned long long ga = (unsigned long long)(uintptr_t)Wsrc[o];
      u32x4 g0;
      g0[0] = 1u;                                          // count=1
      g0[1] = lds_addr;                                    // lds_addr
      g0[2] = (unsigned)(ga & 0xFFFFFFFFu);                // global_addr lo
      g0[3] = (unsigned)((ga >> 32) & 0x01FFFFFFu) | (2u << 30); // hi | type=2
      i32x8 g1;
      g1[0] = (int)(1u << 16);                             // data_size=1 (2B)
      g1[1] = (int)((n & 0xFFFFu) << 16);                  // tensor_dim0 lo
      g1[2] = (int)((n >> 16) | (1u << 16));               // dim0 hi | tensor_dim1=1
      g1[3] = (int)((n & 0xFFFFu) << 16);                  // dim1 hi=0 | tile_dim0=n
      g1[4] = 1;                                           // tile_dim1=1
      g1[5] = (int)n;                                      // tensor_dim0_stride lo
      g1[6] = 0;
      g1[7] = 0;
      i32x4 gz4 = {0, 0, 0, 0};
      i32x8 gz8 = {0, 0, 0, 0, 0, 0, 0, 0};
      __builtin_amdgcn_tensor_load_to_lds(g0, g1, gz4, gz4, gz8, 0);
    }
    __builtin_amdgcn_s_wait_tensorcnt(0);
  }
  __syncthreads();
#else
  {
    const int chunks = NOUT * (SLAB_H / 8);        // uint4 = 8 halves
    for (int i = threadIdx.x; i < chunks; i += blockDim.x) {
      int o = i / (SLAB_H / 8);
      int r = i - o * (SLAB_H / 8);
      ((uint4*)(sW + (size_t)o * SLAB_H))[r] = ((const uint4*)Wsrc[o])[r];
    }
    __syncthreads();
  }
#endif

  const int lane    = threadIdx.x & 31;
  const int wave    = threadIdx.x >> 5;
  const int halfSel = lane >> 4;                   // 0: lanes 0-15, 1: 16-31
  const int lm      = lane & 15;
  const int nTiles  = (nNodes + 15) >> 4;
  const int stride  = gridDim.x * (blockDim.x >> 5);

  const float* Aptr[3] = {A0, A1, A2};
  const float* bptr[2] = {b0, b1};
  float*       optr[2] = {O0, O1};

  for (int tile = blockIdx.x * (blockDim.x >> 5) + wave; tile < nTiles; tile += stride) {
    const int m0   = tile * 16;
    const int rowm = m0 + lm;
    const bool rowok = rowm < nNodes;

    v8f acc[NOUT][4];
#pragma unroll
    for (int o = 0; o < NOUT; ++o)
#pragma unroll
      for (int ct = 0; ct < 4; ++ct) acc[o][ct] = (v8f)0.f;

#pragma unroll
    for (int km = 0; km < 3; ++km) {
      const float* A = Aptr[km];
      for (int kt = 0; kt < kTiles; ++kt) {
        // A fragment: 16x32 f16, contiguous K per lane half (same order as B)
        //   lanes 0-15 : K = kb..kb+15   (kb = kt*32)
        //   lanes 16-31: K = kb+16..kb+31
        const int kb = kt * 32 + halfSel * 16;
        float4 f0 = {0.f, 0.f, 0.f, 0.f}, f1 = f0, g0 = f0, g1 = f0;
        if (rowok) {
          const float* base = A + (size_t)rowm * Kdim;
          if (kb < Kdim)      { f0 = *(const float4*)(base + kb);      f1 = *(const float4*)(base + kb + 4); }
          if (kb + 8 < Kdim)  { g0 = *(const float4*)(base + kb + 8);  g1 = *(const float4*)(base + kb + 12); }
        }
        v16h a;
        a[0]  = (_Float16)f0.x; a[1]  = (_Float16)f0.y; a[2]  = (_Float16)f0.z; a[3]  = (_Float16)f0.w;
        a[4]  = (_Float16)f1.x; a[5]  = (_Float16)f1.y; a[6]  = (_Float16)f1.z; a[7]  = (_Float16)f1.w;
        a[8]  = (_Float16)g0.x; a[9]  = (_Float16)g0.y; a[10] = (_Float16)g0.z; a[11] = (_Float16)g0.w;
        a[12] = (_Float16)g1.x; a[13] = (_Float16)g1.y; a[14] = (_Float16)g1.z; a[15] = (_Float16)g1.w;

        const int koff = kt * 32 + halfSel * 16;   // 16 contiguous K halves
#pragma unroll
        for (int o = 0; o < NOUT; ++o) {
          const _Float16* Wm = sW + ((size_t)(o * 3 + km)) * 4096;
#pragma unroll
          for (int ct = 0; ct < 4; ++ct) {
            const int colB = ct * 16 + lm;
            const v16h b = *(const v16h*)(Wm + colB * 64 + koff);   // ds_load
            acc[o][ct] = __builtin_amdgcn_wmma_f32_16x16x32_f16(
                false, a, false, b, (short)0, acc[o][ct], false, false);
          }
        }
      }
    }

    // store: lane = column, VGPR j = row (lanes>=16: rows 8..15)
#pragma unroll
    for (int o = 0; o < NOUT; ++o) {
#pragma unroll
      for (int ct = 0; ct < 4; ++ct) {
        const int colD = ct * 16 + lm;
        const float bv = bptr[o][colD];
        float* op = optr[o];
#pragma unroll
        for (int j = 0; j < 8; ++j) {
          const int rowD = m0 + halfSel * 8 + j;
          if (rowD < nNodes) {
            const size_t idx = (size_t)rowD * HIDF + colD;
            float v = acc[o][ct][j] + bv;
            if (accum) v += op[idx];
            op[idx] = v;
          }
        }
      }
    }
  }
}

// ------------------------------ GRU gates ----------------------------------

__global__ void gates_kernel(float* __restrict__ gz, const float* __restrict__ gr,
                             const float* __restrict__ h, float* __restrict__ hr, int n) {
  int i = blockIdx.x * blockDim.x + threadIdx.x;
  if (i >= n) return;
  float z = 1.f / (1.f + __expf(-gz[i]));
  gz[i] = z;                                  // keep Z for the update step
  float r = 1.f / (1.f + __expf(-gr[i]));
  hr[i] = h[i] * r;
}

__global__ void update_kernel(float* __restrict__ h, const float* __restrict__ gz,
                              const float* __restrict__ gh, int n) {
  int i = blockIdx.x * blockDim.x + threadIdx.x;
  if (i >= n) return;
  float z  = gz[i];
  float ht = tanhf(gh[i]);
  h[i] = z * h[i] + (1.f - z) * ht;
}

// ------------------------------- output head -------------------------------

__global__ void head_kernel(const float* __restrict__ H,
                            const float* __restrict__ muW, const float* __restrict__ mub,
                            const float* __restrict__ sgW, const float* __restrict__ sgb,
                            float* __restrict__ muo, float* __restrict__ sgo, int nNodes) {
  int gid = blockIdx.x * blockDim.x + threadIdx.x;     // nNodes*2
  if (gid >= nNodes * 2) return;
  int node = gid >> 1, j = gid & 1;
  const float* h = H + (size_t)node * HIDF;
  float am = 0.f, as = 0.f;
#pragma unroll 8
  for (int k = 0; k < HIDF; ++k) {
    float hv = h[k];
    am += hv * muW[k * 2 + j];
    as += hv * sgW[k * 2 + j];
  }
  muo[gid] = am + mub[j];
  float x = as + sgb[j];
  sgo[gid] = (x > 20.f) ? x : log1pf(__expf(x));       // softplus
}

// ------------------------------ host driver --------------------------------

static void run_cell(const float* X, int fin, float* H,
                     const _Float16* Wl,
                     const float* bxz, const float* bxr, const float* bxh,
                     const float* bhz, const float* bhr, const float* bhh,
                     const int* ei, const float* nrm,
                     float* P1, float* P2, float* Gz, float* Gr, float* Gh, float* HR,
                     hipStream_t stream) {
  const int N = N_NODES, E = N_EDGES;
  const size_t slab = (size_t)SLAB_H;
  const int kTx = (fin + 31) / 32;
  const size_t lds2 = 2 * SLAB_H * sizeof(_Float16);   // 48 KB
  const size_t lds1 = 1 * SLAB_H * sizeof(_Float16);   // 24 KB

  // ---- X side: Tx0=X, Tx1=prop(X), Tx2=2*prop(Tx1)-X ; gates xz/xr/xh ----
  {
    const int pairs = fin >> 1;
    const int pb = (int)(((long long)E * pairs + 255) / 256);
    (void)hipMemsetAsync(P1, 0, (size_t)N * fin * 4, stream);
    prop_kernel<<<pb, 256, 0, stream>>>(X, ei, nrm, P1, E, fin);
    (void)hipMemsetAsync(P2, 0, (size_t)N * fin * 4, stream);
    prop_kernel<<<pb, 256, 0, stream>>>(P1, ei, nrm, P2, E, fin);
    fixup_kernel<<<cdiv(N * fin, 256), 256, 0, stream>>>(P2, X, N * fin);
    cheb_gemm_kernel<2><<<GEMM_BLOCKS, 256, lds2, stream>>>(
        X, P1, P2, Wl + 0 * slab, Wl + 1 * slab, bxz, bxr, Gz, Gr, N, fin, kTx, 0);
    cheb_gemm_kernel<1><<<GEMM_BLOCKS, 256, lds1, stream>>>(
        X, P1, P2, Wl + 2 * slab, nullptr, bxh, nullptr, Gh, nullptr, N, fin, kTx, 0);
  }

  // ---- H side: gates hz/hr (accumulate) ----
  {
    const int pb = (int)(((long long)E * 32 + 255) / 256);
    (void)hipMemsetAsync(P1, 0, (size_t)N * HIDF * 4, stream);
    prop_kernel<<<pb, 256, 0, stream>>>(H, ei, nrm, P1, E, HIDF);
    (void)hipMemsetAsync(P2, 0, (size_t)N * HIDF * 4, stream);
    prop_kernel<<<pb, 256, 0, stream>>>(P1, ei, nrm, P2, E, HIDF);
    fixup_kernel<<<cdiv(N * HIDF, 256), 256, 0, stream>>>(P2, H, N * HIDF);
    cheb_gemm_kernel<2><<<GEMM_BLOCKS, 256, lds2, stream>>>(
        H, P1, P2, Wl + 3 * slab, Wl + 4 * slab, bhz, bhr, Gz, Gr, N, HIDF, 2, 1);
  }

  // ---- gates: Z=sigmoid(Gz) (in place), HR = H * sigmoid(Gr) ----
  gates_kernel<<<cdiv(N * HIDF, 256), 256, 0, stream>>>(Gz, Gr, H, HR, N * HIDF);

  // ---- candidate: Gh += cheb(HR, hh) ----
  {
    const int pb = (int)(((long long)E * 32 + 255) / 256);
    (void)hipMemsetAsync(P1, 0, (size_t)N * HIDF * 4, stream);
    prop_kernel<<<pb, 256, 0, stream>>>(HR, ei, nrm, P1, E, HIDF);
    (void)hipMemsetAsync(P2, 0, (size_t)N * HIDF * 4, stream);
    prop_kernel<<<pb, 256, 0, stream>>>(P1, ei, nrm, P2, E, HIDF);
    fixup_kernel<<<cdiv(N * HIDF, 256), 256, 0, stream>>>(P2, HR, N * HIDF);
    cheb_gemm_kernel<1><<<GEMM_BLOCKS, 256, lds1, stream>>>(
        HR, P1, P2, Wl + 5 * slab, nullptr, bhh, nullptr, Gh, nullptr, N, HIDF, 2, 1);
  }

  // ---- H = Z*H + (1-Z)*tanh(Gh) ----
  update_kernel<<<cdiv(N * HIDF, 256), 256, 0, stream>>>(H, Gz, Gh, N * HIDF);
}

extern "C" void kernel_launch(void* const* d_in, const int* in_sizes, int n_in,
                              void* d_out, int out_size, void* d_ws, size_t ws_size,
                              hipStream_t stream) {
  const int N = N_NODES, E = N_EDGES, B = N_B, T = N_T;

  // Pytree leaf order (dict keys sorted): in_tensor, edge_index, edge_weight,
  // layers[0]{hh.W,hh.b,hr.W,hr.b,hz.W,hz.b,xh.W,xh.b,xr.W,xr.b,xz.W,xz.b},
  // layers[1]{...}, mu_W, mu_b, sig_W, sig_b
  const float* in_tensor = (const float*)d_in[0];
  const int*   ei        = (const int*)d_in[1];
  const float* ew        = (const float*)d_in[2];
  const float* muW       = (const float*)d_in[27];
  const float* mub       = (const float*)d_in[28];
  const float* sgW       = (const float*)d_in[29];
  const float* sgb       = (const float*)d_in[30];

  // ---- workspace carve (256B aligned) ----
  size_t off = 0;
  auto carve = [&](size_t bytes) -> void* {
    void* p = (char*)d_ws + off;
    off += (bytes + 255) & ~(size_t)255;
    return p;
  };
  float*    deg = (float*)carve((size_t)N * 4);
  float*    nrm = (float*)carve((size_t)E * 4);
  _Float16* Wh  = (_Float16*)carve((size_t)2 * 6 * SLAB_H * sizeof(_Float16));
  float*    H0  = (float*)carve((size_t)N * HIDF * 4);
  float*    H1  = (float*)carve((size_t)N * HIDF * 4);
  float*    P1  = (float*)carve((size_t)N * HIDF * 4);
  float*    P2  = (float*)carve((size_t)N * HIDF * 4);
  float*    Gz  = (float*)carve((size_t)N * HIDF * 4);
  float*    Gr  = (float*)carve((size_t)N * HIDF * 4);
  float*    Gh  = (float*)carve((size_t)N * HIDF * 4);
  float*    HR  = (float*)carve((size_t)N * HIDF * 4);

  // ---- edge normalization ----
  (void)hipMemsetAsync(deg, 0, (size_t)N * 4, stream);
  deg_kernel<<<cdiv(E, 256), 256, 0, stream>>>(ei, ew, deg, E);
  norm_kernel<<<cdiv(E, 256), 256, 0, stream>>>(ei, ew, deg, nrm, E);

  // ---- weight conversion f32 -> f16 WMMA-B layout ----
  // slot order per layer: 0=xz 1=xr 2=xh 3=hz 4=hr 5=hh
  for (int l = 0; l < 2; ++l) {
    const int base = 3 + 12 * l;
    const int wsrc[6] = {base + 10, base + 8, base + 6, base + 4, base + 2, base + 0};
    for (int m = 0; m < 6; ++m) {
      const int fin = (m < 3 && l == 0) ? 8 : 64;
      prep_w_kernel<<<cdiv(3 * 64 * 64, 256), 256, 0, stream>>>(
          (const float*)d_in[wsrc[m]], Wh + ((size_t)l * 6 + m) * SLAB_H, fin);
    }
  }

  float* mu_out = (float*)d_out;                     // [B][N][2]
  float* sg_out = (float*)d_out + (size_t)B * N * 2; // [B][N][2]

  for (int b = 0; b < B; ++b) {
    (void)hipMemsetAsync(H0, 0, (size_t)N * HIDF * 4, stream);
    (void)hipMemsetAsync(H1, 0, (size_t)N * HIDF * 4, stream);
    for (int t = 0; t < T; ++t) {
      // layer 0: X = in_tensor[b,t] (fin=8), state H0
      {
        const int base = 3;
        const float* X = in_tensor + ((size_t)b * T + t) * N * 8;
        run_cell(X, 8, H0, Wh + (size_t)0 * 6 * SLAB_H,
                 (const float*)d_in[base + 11], (const float*)d_in[base + 9],
                 (const float*)d_in[base + 7],  (const float*)d_in[base + 5],
                 (const float*)d_in[base + 3],  (const float*)d_in[base + 1],
                 ei, nrm, P1, P2, Gz, Gr, Gh, HR, stream);
      }
      // layer 1: X = H0 (fin=64), state H1
      {
        const int base = 15;
        run_cell(H0, HIDF, H1, Wh + (size_t)1 * 6 * SLAB_H,
                 (const float*)d_in[base + 11], (const float*)d_in[base + 9],
                 (const float*)d_in[base + 7],  (const float*)d_in[base + 5],
                 (const float*)d_in[base + 3],  (const float*)d_in[base + 1],
                 ei, nrm, P1, P2, Gz, Gr, Gh, HR, stream);
      }
    }
    head_kernel<<<cdiv(N * 2, 256), 256, 0, stream>>>(
        H1, muW, mub, sgW, sgb,
        mu_out + (size_t)b * N * 2, sg_out + (size_t)b * N * 2, N);
  }
}
